// GRUDecoder_58016418234920
// MI455X (gfx1250) — matrix operand order repaired and checked by reading, not compiled, not run on hardware
//
#include <hip/hip_runtime.h>
#include <hip/hip_bf16.h>

typedef __attribute__((ext_vector_type(16))) _Float16 v16h;
typedef __attribute__((ext_vector_type(8)))  float    v8f;

#define P_PARTS 512
#define I_DIM   256
#define G3      192   // 3*H gates
#define H_DIM   64
#define H1_DIM  128
#define H2_DIM  64
#define H3_DIM  32
#define O_DIM   16

__device__ __forceinline__ float sigmoidf_(float x) {
    return 1.0f / (1.0f + __expf(-x));
}

// y[rows] = W[rows x cols] @ x[cols], W row-major f32 in global, x f16 in LDS,
// y f32 in LDS. rows % 16 == 0, cols % 32 == 0. One 16-row tile per WMMA chain;
// tiles distributed round-robin over the block's waves.
__device__ __forceinline__ void wmma_matvec(const float* __restrict__ Wg,
                                            int rows, int cols,
                                            const _Float16* xlds, float* ylds,
                                            int wave, int nwaves, int lane)
{
    const int colsel = (lane & 16) ? 8 : 0;    // A: hi half-wave starts at K+8
    const int ksel   = (lane & 16) ? 8 : 0;    // B: hi half-wave starts at K+8
    const int mrow   = lane & 15;
    const int ntiles = rows >> 4;
    for (int t = wave; t < ntiles; t += nwaves) {
        v8f c = {};
        const float* wrow = Wg + (size_t)(t * 16 + mrow) * cols;
        for (int k0 = 0; k0 < cols; k0 += 32) {
            // A tile: this lane's row, K = k0+colsel+[0..7] and k0+colsel+16+[0..7]
            const float4* wp = (const float4*)(wrow + k0 + colsel);
            float4 f0 = wp[0], f1 = wp[1], f2 = wp[4], f3 = wp[5];
            v16h a;
            a[0]  = (_Float16)f0.x; a[1]  = (_Float16)f0.y;
            a[2]  = (_Float16)f0.z; a[3]  = (_Float16)f0.w;
            a[4]  = (_Float16)f1.x; a[5]  = (_Float16)f1.y;
            a[6]  = (_Float16)f1.z; a[7]  = (_Float16)f1.w;
            a[8]  = (_Float16)f2.x; a[9]  = (_Float16)f2.y;
            a[10] = (_Float16)f2.z; a[11] = (_Float16)f2.w;
            a[12] = (_Float16)f3.x; a[13] = (_Float16)f3.y;
            a[14] = (_Float16)f3.z; a[15] = (_Float16)f3.w;
            // B tile: broadcast activation chunk (all N columns identical)
            typedef __attribute__((ext_vector_type(8))) _Float16 v8h_;
            v8h_ b0 = *(const v8h_*)(xlds + k0 + ksel);
            v8h_ b1 = *(const v8h_*)(xlds + k0 + 16 + ksel);
            v16h b = __builtin_shufflevector(b0, b1, 0,1,2,3,4,5,6,7,8,9,10,11,12,13,14,15);
            c = __builtin_amdgcn_wmma_f32_16x16x32_f16(
                    false, a, false, b, (short)0, c, false, false);
        }
        // Column N=0: lane 0 holds M=0..7, lane 16 holds M=8..15
        if ((lane & 15) == 0)
            *(v8f*)(ylds + t * 16 + ((lane >> 4) << 3)) = c;
    }
}

__global__ void __launch_bounds__(128)
gru_decoder_kernel(const float* __restrict__ z,
                   const float* __restrict__ W_ih, const float* __restrict__ b_ih,
                   const float* __restrict__ b_hh,
                   const float* __restrict__ W1, const float* __restrict__ b1,
                   const float* __restrict__ W2, const float* __restrict__ b2,
                   const float* __restrict__ W3, const float* __restrict__ b3,
                   const float* __restrict__ W4, const float* __restrict__ b4,
                   float* __restrict__ out)
{
    __shared__ __align__(32) _Float16 zlds[I_DIM];
    __shared__ __align__(32) float    gxraw[G3];
    __shared__ __align__(32) _Float16 hlds[H_DIM];
    __shared__ __align__(32) float    a1raw[H1_DIM];
    __shared__ __align__(32) _Float16 a1h[H1_DIM];
    __shared__ __align__(32) float    a2raw[H2_DIM];
    __shared__ __align__(32) _Float16 a2h[H2_DIM];
    __shared__ __align__(32) float    a3raw[H3_DIM];
    __shared__ __align__(32) _Float16 a3h[H3_DIM];
    __shared__ __align__(32) float    outraw[O_DIM];

    const int p    = blockIdx.x;
    const int tid  = threadIdx.x;
    const int lane = tid & 31;
    const int wave = tid >> 5;
    const int nw   = blockDim.x >> 5;   // 4 waves

    // Stage z[p] into LDS as f16 (WMMA B operand source).
    for (int i = tid; i < I_DIM; i += blockDim.x)
        zlds[i] = (_Float16)z[(size_t)p * I_DIM + i];
    __syncthreads();

    // GRU input projection: gxraw = W_ih[p] @ z[p]   (192 x 256 gemv)
    wmma_matvec(W_ih + (size_t)p * G3 * I_DIM, G3, I_DIM, zlds, gxraw, wave, nw, lane);
    __syncthreads();

    // Gate math (h0 = 0 => hh contribution reduces to b_hh).
    if (tid < H_DIM) {
        const int j = tid;
        const float* bi = b_ih + (size_t)p * G3;
        const float* bh = b_hh + (size_t)p * G3;
        float gr = gxraw[j]       + bi[j];
        float gu = gxraw[64 + j]  + bi[64 + j];
        float gn = gxraw[128 + j] + bi[128 + j];
        float r  = sigmoidf_(gr + bh[j]);
        float u  = sigmoidf_(gu + bh[64 + j]);
        float n  = tanhf(gn + r * bh[128 + j]);
        hlds[j]  = (_Float16)((1.0f - u) * n);
    }
    __syncthreads();

    // MLP layer 1: 64 -> 128, ReLU
    wmma_matvec(W1 + (size_t)p * H1_DIM * H_DIM, H1_DIM, H_DIM, hlds, a1raw, wave, nw, lane);
    __syncthreads();
    if (tid < H1_DIM) {
        float v = a1raw[tid] + b1[(size_t)p * H1_DIM + tid];
        a1h[tid] = (_Float16)fmaxf(v, 0.0f);
    }
    __syncthreads();

    // MLP layer 2: 128 -> 64, ReLU
    wmma_matvec(W2 + (size_t)p * H2_DIM * H1_DIM, H2_DIM, H1_DIM, a1h, a2raw, wave, nw, lane);
    __syncthreads();
    if (tid < H2_DIM) {
        float v = a2raw[tid] + b2[(size_t)p * H2_DIM + tid];
        a2h[tid] = (_Float16)fmaxf(v, 0.0f);
    }
    __syncthreads();

    // MLP layer 3: 64 -> 32, ReLU
    wmma_matvec(W3 + (size_t)p * H3_DIM * H2_DIM, H3_DIM, H2_DIM, a2h, a3raw, wave, nw, lane);
    __syncthreads();
    if (tid < H3_DIM) {
        float v = a3raw[tid] + b3[(size_t)p * H3_DIM + tid];
        a3h[tid] = (_Float16)fmaxf(v, 0.0f);
    }
    __syncthreads();

    // Output layer: 32 -> 16 (no activation)
    wmma_matvec(W4 + (size_t)p * O_DIM * H3_DIM, O_DIM, H3_DIM, a3h, outraw, wave, nw, lane);
    __syncthreads();
    if (tid < O_DIM)
        out[(size_t)p * O_DIM + tid] = outraw[tid] + b4[(size_t)p * O_DIM + tid];
}

extern "C" void kernel_launch(void* const* d_in, const int* in_sizes, int n_in,
                              void* d_out, int out_size, void* d_ws, size_t ws_size,
                              hipStream_t stream) {
    (void)in_sizes; (void)n_in; (void)d_ws; (void)ws_size; (void)out_size;
    const float* z    = (const float*)d_in[0];
    // d_in[1] = num_parts (device scalar, value fixed at 512 by the reference)
    const float* W_ih = (const float*)d_in[2];
    const float* b_ih = (const float*)d_in[3];
    // d_in[4] = W_hh — inert (h0 == 0), never read: saves 25 MB of traffic
    const float* b_hh = (const float*)d_in[5];
    const float* W1   = (const float*)d_in[6];
    const float* b1   = (const float*)d_in[7];
    const float* W2   = (const float*)d_in[8];
    const float* b2   = (const float*)d_in[9];
    const float* W3   = (const float*)d_in[10];
    const float* b3   = (const float*)d_in[11];
    const float* W4   = (const float*)d_in[12];
    const float* b4   = (const float*)d_in[13];
    float* out = (float*)d_out;

    dim3 grid(P_PARTS), block(128);
    gru_decoder_kernel<<<grid, block, 0, stream>>>(
        z, W_ih, b_ih, b_hh, W1, b1, W2, b2, W3, b3, W4, b4, out);
}
